// GraphSAGE_14980845929105
// MI455X (gfx1250) — matrix operand order repaired and checked by reading, not compiled, not run on hardware
//
#include <hip/hip_runtime.h>
#include <hip/hip_bf16.h>
#include <math.h>

// ---------------------------------------------------------------------------
// GraphSAGE (3-layer, mean aggregation) for MI455X / gfx1250.
//   layer: h = relu( (agg/deg) @ Wl + b + x @ Wr )
// GEMM runs as one fused K=256 WMMA pass: A = [mean || x], B = [Wl ; Wr].
// A panel staged into LDS with gfx1250 async global->LDS copies (ASYNCcnt),
// B staged as f16 fragments read with ds_load_b128.
// ---------------------------------------------------------------------------

typedef __attribute__((ext_vector_type(16))) _Float16 v16h;
typedef __attribute__((ext_vector_type(8)))  float    v8f;

#define FEAT 128        // feature width of every scatter (layers 1..3 inputs)
#define AROW 132        // padded LDS row stride in floats (132 % 64 = 4 -> no bank conflicts)

// --- gfx1250 async global->LDS copy (tracked by ASYNCcnt) ------------------
__device__ __forceinline__ void async_g2l_b128(unsigned lds_off, const void* gsrc) {
  asm volatile("global_load_async_to_lds_b128 %0, %1, off"
               :: "v"(lds_off), "v"(gsrc)
               : "memory");
}
__device__ __forceinline__ void wait_async0() {
  asm volatile("s_wait_asynccnt 0x0" ::: "memory");
}

// ---------------------------------------------------------------------------
__global__ void zero_kernel(float* __restrict__ p, size_t n) {
  size_t i = (size_t)blockIdx.x * blockDim.x + threadIdx.x;
  size_t stride = (size_t)gridDim.x * blockDim.x;
  for (; i < n; i += stride) p[i] = 0.0f;
}

// degree count: one thread per edge
__global__ void deg_kernel(const int* __restrict__ dst, float* __restrict__ deg, int E) {
  int e = blockIdx.x * blockDim.x + threadIdx.x;
  if (e < E) atomicAdd(&deg[dst[e]], 1.0f);
}

// rdeg = 1 / max(deg, 1)
__global__ void rdeg_kernel(float* __restrict__ deg, int N) {
  int i = blockIdx.x * blockDim.x + threadIdx.x;
  if (i < N) deg[i] = 1.0f / fmaxf(deg[i], 1.0f);
}

// one wave32 per edge; lane handles 4 consecutive features (32*4 = 128).
// agg rows (51 MB) are L2-resident on MI455X (192 MB L2) -> atomics resolve at L2.
__global__ void scatter_kernel(const float* __restrict__ feat,
                               const int*   __restrict__ src,
                               const int*   __restrict__ dst,
                               float*       __restrict__ agg, int E) {
  int wave = (blockIdx.x * blockDim.x + threadIdx.x) >> 5;
  int lane = threadIdx.x & 31;
  if (wave >= E) return;
  int s = src[wave];
  int d = dst[wave];
  // warm L2 for a future edge's source row (gfx1250 global_prefetch_b8)
  int eNext = wave + 64;
  if (eNext < E) {
    int s2 = src[eNext];
    __builtin_prefetch(feat + (size_t)s2 * FEAT + lane * 4, 0, 1);
  }
  const float4 v = *(const float4*)(feat + (size_t)s * FEAT + lane * 4);
  float* base = agg + (size_t)d * FEAT + lane * 4;
  atomicAdd(base + 0, v.x);
  atomicAdd(base + 1, v.y);
  atomicAdd(base + 2, v.z);
  atomicAdd(base + 3, v.w);
}

// ---------------------------------------------------------------------------
// Fused SAGE GEMM: out[n, :] = act( mean[n] @ Wl + x[n] @ Wr + bias )
// 256 threads = 8 waves; wave w computes rows [blk*128 + 16w, +16).
// K = 256 (mean rows 0..127, x rows 128..255), stepped 32 per WMMA.
//  - A panel (agg tile + x tile, f32) staged via async global->LDS b128 copies
//  - weights staged as f16, pre-swizzled into B-fragment order (ds_load_b128)
// ---------------------------------------------------------------------------
template<int NOUT, int NTILES, bool RELU>
__global__ void __launch_bounds__(256)
sage_gemm_kernel(const float* __restrict__ xin,   // [N,128] root features
                 const float* __restrict__ agg,   // [N,128] neighbor sums
                 const float* __restrict__ rdeg,  // [N] 1/max(deg,1)
                 const float* __restrict__ Wl,    // [128,NOUT]
                 const float* __restrict__ Wr,    // [128,NOUT]
                 const float* __restrict__ bias,  // [NOUT]
                 float*       __restrict__ out,   // [N,NOUT]
                 int N) {
  // B fragments: [tile][kk32][lane][16 halfs] -- 32B contiguous per lane
  __shared__ _Float16 sWf[NTILES * 8 * 32 * 16];
  // dynamic LDS: f32 A panels, agg then x, each 128 rows x AROW floats
  extern __shared__ __align__(16) char dynLds[];
  float* sAg = (float*)dynLds;
  float* sAx = sAg + 128 * AROW;

  const int tid = threadIdx.x;
  const long rowBase0 = (long)blockIdx.x * 128;

  // ---- kick off async staging of the A panel (128 rows x 2 panels) --------
  // 128 floats/row/panel = 32 b128 chunks; 8192 chunks / 256 threads = 32 each
  for (int c = tid; c < 128 * 32 * 2; c += 256) {
    const int panel = c >> 12;          // 0: agg, 1: x
    const int r     = (c >> 5) & 127;   // row within block
    const int q     = c & 31;           // 16B chunk within row
    long rowg = rowBase0 + r;
    if (rowg >= N) rowg = N - 1;        // clamp; masked to zero at use
    const float* g = (panel ? xin : agg) + rowg * FEAT + q * 4;
    float*       l = (panel ? sAx : sAg) + r * AROW + q * 4;
    async_g2l_b128((unsigned)(uintptr_t)l, g);
  }

  // ---- stage + swizzle weights into fragment order (overlaps async) -------
  for (int idx = tid; idx < 256 * NTILES * 16; idx += 256) {
    int k = idx / (NTILES * 16);          // 0..255  (K index: Wl then Wr)
    int n = idx % (NTILES * 16);          // padded column
    float w = 0.0f;
    if (n < NOUT) w = (k < 128) ? Wl[k * NOUT + n] : Wr[(k - 128) * NOUT + n];
    int t    = n >> 4;
    int ko   = k & 31;                    // position within 32-wide K block
    int ln   = (((ko >> 3) & 1) << 4) + (n & 15);   // half*16 + col-in-tile
    int frag = t * 8 + (k >> 5);
    int e    = (ko & 7) + ((ko >> 4) << 3);
    sWf[(frag * 32 + ln) * 16 + e] = (_Float16)w;
  }

  wait_async0();        // this wave's async copies landed in LDS
  __syncthreads();      // all waves' copies + weight swizzle visible

  const int  lane  = tid & 31;
  const int  wv    = tid >> 5;
  const int  half  = lane >> 4;
  const int  mloc  = lane & 15;
  const long rowBase = rowBase0 + wv * 16;
  const long row     = rowBase + mloc;          // A row owned by this lane
  const bool rvalid  = row < (long)N;
  const long rowc    = rvalid ? row : (long)(N - 1);
  const float vmask  = rvalid ? 1.0f : 0.0f;
  const float rd     = rdeg[rowc] * vmask;
  const int   rloc   = wv * 16 + mloc;          // LDS row of this lane

  v8f acc[NTILES];
  v8f zacc = {};
#pragma unroll
  for (int t = 0; t < NTILES; ++t) acc[t] = zacc;

  for (int kk32 = 0; kk32 < 8; ++kk32) {
    const bool   isMean = kk32 < 4;               // K 0..127 = mean, 128..255 = x
    const float* rowp   = (isMean ? sAg : sAx) + rloc * AROW + (kk32 & 3) * 32;
    const float  sc     = isMean ? rd : vmask;

    // A fragment, ISA 16-bit 16x32 layout:
    //   lanes 0-15 : K = kk+[0..7]  and kk+[16..23]
    //   lanes16-31 : K = kk+[8..15] and kk+[24..31]
    // 16B-aligned by construction (row stride 528B, runs at 32B boundaries)
    const float4 p0 = *(const float4*)(rowp + half * 8);
    const float4 p1 = *(const float4*)(rowp + half * 8 + 4);
    const float4 p2 = *(const float4*)(rowp + 16 + half * 8);
    const float4 p3 = *(const float4*)(rowp + 16 + half * 8 + 4);

    v16h a;
    a[0]  = (_Float16)(p0.x * sc);  a[1]  = (_Float16)(p0.y * sc);
    a[2]  = (_Float16)(p0.z * sc);  a[3]  = (_Float16)(p0.w * sc);
    a[4]  = (_Float16)(p1.x * sc);  a[5]  = (_Float16)(p1.y * sc);
    a[6]  = (_Float16)(p1.z * sc);  a[7]  = (_Float16)(p1.w * sc);
    a[8]  = (_Float16)(p2.x * sc);  a[9]  = (_Float16)(p2.y * sc);
    a[10] = (_Float16)(p2.z * sc);  a[11] = (_Float16)(p2.w * sc);
    a[12] = (_Float16)(p3.x * sc);  a[13] = (_Float16)(p3.y * sc);
    a[14] = (_Float16)(p3.z * sc);  a[15] = (_Float16)(p3.w * sc);

#pragma unroll
    for (int t = 0; t < NTILES; ++t) {
      const v16h b = *(const v16h*)&sWf[((t * 8 + kk32) * 32 + lane) * 16];
      acc[t] = __builtin_amdgcn_wmma_f32_16x16x32_f16(
          /*neg_a=*/false, a, /*neg_b=*/false, b,
          /*c_mod=*/(short)0, acc[t], /*reuse_a=*/false, /*reuse_b=*/false);
    }
  }

  // ---- epilogue: C layout VGPR v -> M = v + 8*half, N = mloc --------------
#pragma unroll
  for (int t = 0; t < NTILES; ++t) {
    const int col = t * 16 + mloc;
#pragma unroll
    for (int v = 0; v < 8; ++v) {
      const long orow = rowBase + v + half * 8;
      if (orow < (long)N && col < NOUT) {
        float val = acc[t][v] + bias[col];
        if (RELU) val = fmaxf(val, 0.0f);
        out[orow * NOUT + col] = val;
      }
    }
  }
}

// ---------------------------------------------------------------------------
// log_softmax over 40 classes: one wave per row, shuffle reductions.
// ---------------------------------------------------------------------------
__global__ void logsoftmax_kernel(const float* __restrict__ logits,
                                  float* __restrict__ out, int N) {
  int wave = (blockIdx.x * blockDim.x + threadIdx.x) >> 5;
  int lane = threadIdx.x & 31;
  if (wave >= N) return;
  const float* rowp = logits + (long)wave * 40;
  float v0 = rowp[lane];                                   // lane < 32 < 40
  float v1 = (lane + 32 < 40) ? rowp[lane + 32] : -INFINITY;
  float m = fmaxf(v0, v1);
  for (int off = 16; off; off >>= 1) m = fmaxf(m, __shfl_xor(m, off, 32));
  float s = expf(v0 - m) + ((lane + 32 < 40) ? expf(v1 - m) : 0.0f);
  for (int off = 16; off; off >>= 1) s += __shfl_xor(s, off, 32);
  float lse = m + logf(s);
  out[(long)wave * 40 + lane] = v0 - lse;
  if (lane + 32 < 40) out[(long)wave * 40 + lane + 32] = v1 - lse;
}

// ---------------------------------------------------------------------------
extern "C" void kernel_launch(void* const* d_in, const int* in_sizes, int n_in,
                              void* d_out, int out_size, void* d_ws, size_t ws_size,
                              hipStream_t stream) {
  const float* x   = (const float*)d_in[0];
  const int*   ei  = (const int*)  d_in[1];
  const float* W1l = (const float*)d_in[2];
  const float* b1  = (const float*)d_in[3];
  const float* W1r = (const float*)d_in[4];
  const float* W2l = (const float*)d_in[5];
  const float* b2  = (const float*)d_in[6];
  const float* W2r = (const float*)d_in[7];
  const float* W3l = (const float*)d_in[8];
  const float* b3  = (const float*)d_in[9];
  const float* W3r = (const float*)d_in[10];

  const int N = in_sizes[0] / FEAT;
  const int E = in_sizes[1] / 2;
  const int* src = ei;
  const int* dst = ei + E;

  // workspace layout (floats)
  float* ws     = (float*)d_ws;
  float* rdeg   = ws;                                   // N (holds deg then 1/deg)
  float* agg    = ws + (((size_t)N + 255) & ~(size_t)255);
  float* h1     = agg + (size_t)N * FEAT;               // layer-1 output
  float* logits = h1  + (size_t)N * FEAT;               // N*40

  float* outLsm = (float*)d_out;                        // [N,40]
  float* h2     = (float*)d_out + (size_t)N * 40;       // embeddings [N,128]

  const int thr = 256;
  const int zeroBlocks = 4096;
  const int degBlocks  = (E + thr - 1) / thr;
  const int nBlocks    = (N + thr - 1) / thr;
  const int scatBlocks = (E + 7) / 8;                   // 8 edges (waves) / block
  const int gemmBlocks = (N + 127) / 128;
  const int lsmBlocks  = (N + 7) / 8;
  const size_t dynBytes = (size_t)2 * 128 * AROW * sizeof(float);  // A panels

  // degree (shared by all layers)
  zero_kernel<<<zeroBlocks, thr, 0, stream>>>(rdeg, (size_t)N);
  deg_kernel<<<degBlocks, thr, 0, stream>>>(dst, rdeg, E);
  rdeg_kernel<<<nBlocks, thr, 0, stream>>>(rdeg, N);

  // ---- layer 1 ----
  zero_kernel<<<zeroBlocks, thr, 0, stream>>>(agg, (size_t)N * FEAT);
  scatter_kernel<<<scatBlocks, thr, 0, stream>>>(x, src, dst, agg, E);
  sage_gemm_kernel<128, 8, true><<<gemmBlocks, thr, dynBytes, stream>>>(
      x, agg, rdeg, W1l, W1r, b1, h1, N);

  // ---- layer 2 (embeddings straight into d_out) ----
  zero_kernel<<<zeroBlocks, thr, 0, stream>>>(agg, (size_t)N * FEAT);
  scatter_kernel<<<scatBlocks, thr, 0, stream>>>(h1, src, dst, agg, E);
  sage_gemm_kernel<128, 8, true><<<gemmBlocks, thr, dynBytes, stream>>>(
      h1, agg, rdeg, W2l, W2r, b2, h2, N);

  // ---- layer 3 (40 classes, padded to 3 WMMA N-tiles) ----
  zero_kernel<<<zeroBlocks, thr, 0, stream>>>(agg, (size_t)N * FEAT);
  scatter_kernel<<<scatBlocks, thr, 0, stream>>>(h2, src, dst, agg, E);
  sage_gemm_kernel<40, 3, false><<<gemmBlocks, thr, dynBytes, stream>>>(
      h2, agg, rdeg, W3l, W3r, b3, logits, N);

  logsoftmax_kernel<<<lsmBlocks, thr, 0, stream>>>(logits, outLsm, N);
}
